// NP_3032246911158
// MI455X (gfx1250) — matrix operand order repaired and checked by reading, not compiled, not run on hardware
//
#include <hip/hip_runtime.h>
#include <hip/hip_bf16.h>
#include <math.h>

typedef __attribute__((ext_vector_type(16))) _Float16 v16h;
typedef __attribute__((ext_vector_type(8)))  float    v8f;

#define NTOK   32768      // B * S = 32 * 1024
#define BATCH  32
#define SEQ    1024
#define HDIM   64

// ---------------------------------------------------------------------------
// One-shot f32 -> f16 weight conversion (64x64 = 4096 elements per matrix)
// ---------------------------------------------------------------------------
__global__ void cvtw_kernel(const float* __restrict__ W, _Float16* __restrict__ Wh, int n)
{
  int i = blockIdx.x * blockDim.x + threadIdx.x;
  if (i < n) Wh[i] = (_Float16)W[i];
}

// ---------------------------------------------------------------------------
// Token-wise GEMM: Y[t,:] = act(X[t,:] @ W(64x64) + bias), W pre-converted f16.
// One wave = 32 tokens (2 M-tiles sharing B registers). K=64 -> 2 WMMAs per
// 16x16 tile; 2 M-tiles x 4 N-tiles -> 16 x v_wmma_f32_16x16x32_f16 per wave.
// ---------------------------------------------------------------------------
__global__ __launch_bounds__(128) void gemm64_kernel(
    const float* __restrict__ X, const _Float16* __restrict__ Wh,
    const float* __restrict__ bias, float* __restrict__ Y,
    int act /*0=none,1=relu*/, int ntok)
{
  const int lane = threadIdx.x & 31;
  const int wave = threadIdx.x >> 5;
  const int tok0 = (blockIdx.x * 4 + wave) * 32;   // 32 tokens per wave
  if (tok0 >= ntok) return;                         // wave-uniform branch

  const int hi = lane >> 4;
  const int m  = lane & 15;

  // ---- B operand: lane = N (lane&15), elem s -> K = 16*(s>>3) + 8*hi + (s&7)
  v16h Bm[2][4];
#pragma unroll
  for (int kt = 0; kt < 2; ++kt) {
#pragma unroll
    for (int nt = 0; nt < 4; ++nt) {
      v16h bb;
#pragma unroll
      for (int s = 0; s < 16; ++s) {
        const int k = kt * 32 + 16 * (s >> 3) + 8 * hi + (s & 7);
        bb[s] = Wh[k * 64 + nt * 16 + m];
      }
      Bm[kt][nt] = bb;
    }
  }

  // ---- A operand: 2 M-tiles, f32 activations converted in-register
  v16h Am[2][2];   // [mt][kt]
#pragma unroll
  for (int mt = 0; mt < 2; ++mt) {
    const float* xp = X + (size_t)(tok0 + mt * 16 + m) * 64;
#pragma unroll
    for (int kt = 0; kt < 2; ++kt) {
      v16h aa;
#pragma unroll
      for (int v = 0; v < 8; ++v) {
        const int g = v >> 2;
        const int k = kt * 32 + 16 * g + 8 * hi + 2 * (v & 3);
        aa[2 * v]     = (_Float16)xp[k];
        aa[2 * v + 1] = (_Float16)xp[k + 1];
      }
      Am[mt][kt] = aa;
    }
  }

  // ---- 16 WMMAs, f32 accumulate
#pragma unroll
  for (int mt = 0; mt < 2; ++mt) {
#pragma unroll
    for (int nt = 0; nt < 4; ++nt) {
      v8f c = {};
      c = __builtin_amdgcn_wmma_f32_16x16x32_f16(false, Am[mt][0], false, Bm[0][nt],
                                                 (short)0, c, false, false);
      c = __builtin_amdgcn_wmma_f32_16x16x32_f16(false, Am[mt][1], false, Bm[1][nt],
                                                 (short)0, c, false, false);
      const int n  = nt * 16 + m;                  // C/D: lane = N
      const float bv = bias ? bias[n] : 0.0f;
#pragma unroll
      for (int r = 0; r < 8; ++r) {                // C/D: elem r -> M = r + 8*hi
        const int mm = tok0 + mt * 16 + r + 8 * hi;
        float val = c[r] + bv;
        if (act == 1) val = fmaxf(val, 0.0f);
        Y[(size_t)mm * 64 + n] = val;
      }
    }
  }
}

// ---------------------------------------------------------------------------
// First MLP layer (fan_in = 1 or 2): rank-<=2 outer product + bias + relu
// ---------------------------------------------------------------------------
__global__ void firstlayer_kernel(const float* __restrict__ X, int din, int xstride,
                                  const float* __restrict__ W, const float* __restrict__ b,
                                  float* __restrict__ Y, int ntok)
{
  int i = blockIdx.x * blockDim.x + threadIdx.x;
  if (i >= ntok * 64) return;
  int t = i >> 6, n = i & 63;
  float acc = b[n];
  for (int c = 0; c < din; ++c) acc += X[(size_t)t * xstride + c] * W[c * 64 + n];
  Y[i] = fmaxf(acc, 0.0f);
}

// ---------------------------------------------------------------------------
// q/k projection (64 -> 8) + cos/sin
// ---------------------------------------------------------------------------
__global__ void qkproj_kernel(const float* __restrict__ X, const float* __restrict__ Wp,
                              float* __restrict__ C, float* __restrict__ S, int ntok)
{
  int i = blockIdx.x * blockDim.x + threadIdx.x;  // (t, h)
  if (i >= ntok * 8) return;
  int t = i >> 3, h = i & 7;
  const float* xr = X + (size_t)t * 64;
  float acc = 0.f;
#pragma unroll 8
  for (int k = 0; k < 64; ++k) acc += xr[k] * Wp[k * 8 + h];
  C[i] = cosf(acc);
  S[i] = sinf(acc);
}

// ---------------------------------------------------------------------------
// Per-batch linear-attention reductions:
//   Mc = CK^T V (8x64), Ms = SK^T V, ckS = sum_k ck, skS = sum_k sk
// ---------------------------------------------------------------------------
__global__ __launch_bounds__(512) void attn_reduce_kernel(
    const float* __restrict__ CK, const float* __restrict__ SK,
    const float* __restrict__ V,
    float* __restrict__ Mc, float* __restrict__ Ms,
    float* __restrict__ ckS, float* __restrict__ skS)
{
  const int b   = blockIdx.x;
  const int tid = threadIdx.x;              // 512 = 8 heads * 64 dims
  const int h = tid >> 6, d = tid & 63;
  const float* ck = CK + (size_t)b * SEQ * 8;
  const float* sk = SK + (size_t)b * SEQ * 8;
  const float* v  = V  + (size_t)b * SEQ * 64;
  float ac = 0.f, as = 0.f;
  for (int k = 0; k < SEQ; ++k) {
    float vk = v[(size_t)k * 64 + d];
    ac += ck[k * 8 + h] * vk;
    as += sk[k * 8 + h] * vk;
  }
  Mc[b * 512 + tid] = ac;                   // [b][h][d]
  Ms[b * 512 + tid] = as;
  if (tid < 8) {
    float sc = 0.f, ss = 0.f;
    for (int k = 0; k < SEQ; ++k) { sc += ck[k * 8 + tid]; ss += sk[k * 8 + tid]; }
    ckS[b * 8 + tid] = sc;
    skS[b * 8 + tid] = ss;
  }
}

// ---------------------------------------------------------------------------
// r[q,d] = (cq.Mc[:,d] + sq.Ms[:,d])/8 / ((cq.ckS + sq.skS)/8 + EPS)
// ---------------------------------------------------------------------------
__global__ void attn_combine_kernel(
    const float* __restrict__ CQ, const float* __restrict__ SQ,
    const float* __restrict__ Mc, const float* __restrict__ Ms,
    const float* __restrict__ ckS, const float* __restrict__ skS,
    float* __restrict__ R, int total)
{
  int i = blockIdx.x * blockDim.x + threadIdx.x;
  if (i >= total) return;
  const int d = i & 63;
  const int t = i >> 6;                     // global token
  const int b = t >> 10;                    // SEQ = 1024
  const float* cq = CQ + (size_t)t * 8;
  const float* sq = SQ + (size_t)t * 8;
  float num = 0.f, den = 0.f;
#pragma unroll
  for (int h = 0; h < 8; ++h) {
    num += cq[h] * Mc[b * 512 + h * 64 + d] + sq[h] * Ms[b * 512 + h * 64 + d];
    den += cq[h] * ckS[b * 8 + h] + sq[h] * skS[b * 8 + h];
  }
  R[i] = (num * 0.125f) / (den * 0.125f + 1e-6f);
}

// ---------------------------------------------------------------------------
// LayerNorm over 64 features of (A + B): one wave32 per token, 2 feats/lane
// ---------------------------------------------------------------------------
__global__ __launch_bounds__(128) void ln_kernel(
    const float* __restrict__ A, const float* __restrict__ Badd,
    const float* __restrict__ g, const float* __restrict__ be,
    float* __restrict__ Y, int ntok)
{
  const int lane = threadIdx.x & 31;
  const int wave = threadIdx.x >> 5;
  const int t = blockIdx.x * 4 + wave;
  if (t >= ntok) return;
  const size_t base = (size_t)t * 64;
  float x0 = A[base + lane]      + (Badd ? Badd[base + lane]      : 0.f);
  float x1 = A[base + 32 + lane] + (Badd ? Badd[base + 32 + lane] : 0.f);
  float s = x0 + x1;
#pragma unroll
  for (int o = 16; o; o >>= 1) s += __shfl_xor(s, o, 32);
  const float mean = s * (1.0f / 64.0f);
  const float d0 = x0 - mean, d1 = x1 - mean;
  float vs = d0 * d0 + d1 * d1;
#pragma unroll
  for (int o = 16; o; o >>= 1) vs += __shfl_xor(vs, o, 32);
  const float inv = rsqrtf(vs * (1.0f / 64.0f) + 1e-5f);
  Y[base + lane]      = d0 * inv * g[lane]      + be[lane];
  Y[base + 32 + lane] = d1 * inv * g[lane + 32] + be[lane + 32];
}

// ---------------------------------------------------------------------------
// Decoder last layer (64 -> 2): mu, softplus(sigma)
// ---------------------------------------------------------------------------
__global__ void declast_kernel(const float* __restrict__ X, const float* __restrict__ W,
                               const float* __restrict__ b, float* __restrict__ out, int ntok)
{
  int t = blockIdx.x * blockDim.x + threadIdx.x;
  if (t >= ntok) return;
  const float* xr = X + (size_t)t * 64;
  float a0 = b[0], a1 = b[1];
#pragma unroll 8
  for (int k = 0; k < 64; ++k) { a0 += xr[k] * W[k * 2]; a1 += xr[k] * W[k * 2 + 1]; }
  out[t] = a0;
  out[ntok + t] = (a1 > 20.f) ? a1 : log1pf(expf(a1));
}

// ---------------------------------------------------------------------------
extern "C" void kernel_launch(void* const* d_in, const int* in_sizes, int n_in,
                              void* d_out, int out_size, void* d_ws, size_t ws_size,
                              hipStream_t stream)
{
  auto F = [&](int i) { return (const float*)d_in[i]; };
  const float* xy  = F(0);
  const float* qin = F(1);

  // cross blocks: 15 leaves each (Wk,Wq,Wv,finalW,finalb,l1g,l1b,l2g,l2b,m0W,m0b,m1W,m1b,m2W,m2b)
  struct CrossP { const float *Wk,*Wq,*Wv,*fW,*fb,*l1g,*l1b,*l2g,*l2b,*m0W,*m0b,*m1W,*m1b,*m2W,*m2b; };
  CrossP cr[4];
  for (int p = 0; p < 4; ++p) {
    int s = 2 + 15 * p;
    cr[p] = { F(s+0),F(s+1),F(s+2),F(s+3),F(s+4),F(s+5),F(s+6),F(s+7),
              F(s+8),F(s+9),F(s+10),F(s+11),F(s+12),F(s+13),F(s+14) };
  }
  const float *decW0=F(62),*decb0=F(63),*decW1=F(64),*decb1=F(65),*decW2=F(66),*decb2=F(67);
  const float *kcW0 =F(68),*kcb0 =F(69),*kcW1 =F(70),*kcb1 =F(71);
  const float *qW0  =F(72),*qb0  =F(73),*qW1  =F(74),*qb1  =F(75);
  const float *xyW0 =F(76),*xyb0 =F(77),*xyW1 =F(78),*xyb1 =F(79);

  // workspace layout (floats)
  float* ws = (float*)d_ws;
  const size_t FB = (size_t)NTOK * 64;     // 2M floats per activation buffer
  float* ENC = ws + 0 * FB;
  float* KC  = ws + 1 * FB;
  float* Q0  = ws + 2 * FB;
  float* QE  = ws + 3 * FB;
  float* V   = ws + 4 * FB;                // freed after attn_reduce -> reused as temp3
  float* T1  = ws + 5 * FB;
  float* T2  = ws + 6 * FB;
  float* CQ  = ws + 7 * FB;
  float* SQ  = CQ + (size_t)NTOK * 8;
  float* CK  = SQ + (size_t)NTOK * 8;
  float* SK  = CK + (size_t)NTOK * 8;
  float* Mc  = SK + (size_t)NTOK * 8;
  float* Ms  = Mc + BATCH * 512;
  float* ckS = Ms + BATCH * 512;
  float* skS = ckS + BATCH * 8;

  // f16 weight arena (32B-aligned: follows float buffers at 64MB offset)
  _Float16* WH = (_Float16*)(ws + (size_t)16 * 1024 * 1024);
  // slots: 0=xyW1 1=kcW1 2=qW1; per cross p: 3+5p..7+5p = Wv,fW,m0,m1,m2; 23=dec0 24=dec1
  const float* wsrc[25];
  wsrc[0] = xyW1; wsrc[1] = kcW1; wsrc[2] = qW1;
  for (int p = 0; p < 4; ++p) {
    wsrc[3 + 5*p] = cr[p].Wv;  wsrc[4 + 5*p] = cr[p].fW;
    wsrc[5 + 5*p] = cr[p].m0W; wsrc[6 + 5*p] = cr[p].m1W; wsrc[7 + 5*p] = cr[p].m2W;
  }
  wsrc[23] = decW0; wsrc[24] = decW1;
  for (int i = 0; i < 25; ++i)
    cvtw_kernel<<<16, 256, 0, stream>>>(wsrc[i], WH + (size_t)i * 4096, 4096);
  auto WHs = [&](int i) { return (const _Float16*)(WH + (size_t)i * 4096); };

  const int gemmBlocks = NTOK / 128;                   // 4 waves/block, 32 tok/wave
  const int ewBlocks64 = (NTOK * 64 + 255) / 256;
  const int ewBlocks8  = (NTOK * 8 + 255) / 256;
  const int lnBlocks   = (NTOK + 3) / 4;

  // ---- encoders -----------------------------------------------------------
  firstlayer_kernel<<<ewBlocks64, 256, 0, stream>>>(xy, 2, 2, xyW0, xyb0, T1, NTOK);
  gemm64_kernel<<<gemmBlocks, 128, 0, stream>>>(T1, WHs(0), xyb1, ENC, 0, NTOK);
  firstlayer_kernel<<<ewBlocks64, 256, 0, stream>>>(xy, 1, 2, kcW0, kcb0, T1, NTOK);
  gemm64_kernel<<<gemmBlocks, 128, 0, stream>>>(T1, WHs(1), kcb1, KC, 0, NTOK);
  firstlayer_kernel<<<ewBlocks64, 256, 0, stream>>>(qin, 1, 1, qW0, qb0, T1, NTOK);
  gemm64_kernel<<<gemmBlocks, 128, 0, stream>>>(T1, WHs(2), qb1, Q0, 0, NTOK);

  // ---- 4 cross-attention blocks (linear-attention factorization) ----------
  for (int p = 0; p < 4; ++p) {
    const float* resid = (p == 0) ? Q0 : QE;
    qkproj_kernel<<<ewBlocks8, 256, 0, stream>>>(KC, cr[p].Wk, CK, SK, NTOK);
    qkproj_kernel<<<ewBlocks8, 256, 0, stream>>>(Q0, cr[p].Wq, CQ, SQ, NTOK);
    gemm64_kernel<<<gemmBlocks, 128, 0, stream>>>(ENC, WHs(3 + 5*p), nullptr, V, 0, NTOK);
    attn_reduce_kernel<<<BATCH, 512, 0, stream>>>(CK, SK, V, Mc, Ms, ckS, skS);
    attn_combine_kernel<<<ewBlocks64, 256, 0, stream>>>(CQ, SQ, Mc, Ms, ckS, skS,
                                                        T1, NTOK * 64);
    gemm64_kernel<<<gemmBlocks, 128, 0, stream>>>(T1, WHs(4 + 5*p), cr[p].fb, T2, 0, NTOK);
    ln_kernel<<<lnBlocks, 128, 0, stream>>>(T2, resid, cr[p].l1g, cr[p].l1b, T1, NTOK);
    gemm64_kernel<<<gemmBlocks, 128, 0, stream>>>(T1, WHs(5 + 5*p), cr[p].m0b, T2, 1, NTOK);
    gemm64_kernel<<<gemmBlocks, 128, 0, stream>>>(T2, WHs(6 + 5*p), cr[p].m1b, V,  1, NTOK);
    gemm64_kernel<<<gemmBlocks, 128, 0, stream>>>(V,  WHs(7 + 5*p), cr[p].m2b, T2, 0, NTOK);
    ln_kernel<<<lnBlocks, 128, 0, stream>>>(T1, T2, cr[p].l2g, cr[p].l2b, QE, NTOK);
  }

  // ---- decoder ------------------------------------------------------------
  gemm64_kernel<<<gemmBlocks, 128, 0, stream>>>(QE, WHs(23), decb0, T1, 1, NTOK);
  gemm64_kernel<<<gemmBlocks, 128, 0, stream>>>(T1, WHs(24), decb1, T2, 1, NTOK);
  declast_kernel<<<(NTOK + 255) / 256, 256, 0, stream>>>(T2, decW2, decb2,
                                                         (float*)d_out, NTOK);
}
